// LVSMDecoderOnlyModel_79877801771303
// MI455X (gfx1250) — matrix operand order repaired and checked by reading, not compiled, not run on hardware
//
#include <hip/hip_runtime.h>

#define IMG   256
#define PP    8
#define DM    768
#define NHEAD 16
#define HDIM  48
#define DFFN  3072
#define NLAY  6
#define NTOK  1024
#define SEQ   3072

typedef __attribute__((ext_vector_type(16))) __bf16 v16bf;
typedef __attribute__((ext_vector_type(8)))  float  v8f;
typedef __attribute__((ext_vector_type(4)))  unsigned int u32x4;
typedef __attribute__((ext_vector_type(4)))  int i32x4;
typedef __attribute__((ext_vector_type(8)))  int i32x8;

union FragBF {
  v16bf v;
  uint4 q[2];
  unsigned short u[16];
};

__device__ __forceinline__ unsigned short f2bf(float f) {
  unsigned int u = __float_as_uint(f);
  unsigned int r = 0x7FFFu + ((u >> 16) & 1u);
  return (unsigned short)((u + r) >> 16);
}

__device__ __forceinline__ v8f vzero8() {
  v8f z;
#pragma unroll
  for (int i = 0; i < 8; ++i) z[i] = 0.f;
  return z;
}

__device__ __forceinline__ v8f wmma_bf(const FragBF& a, const FragBF& b, v8f c) {
  return __builtin_amdgcn_wmma_f32_16x16x32_bf16(false, a.v, false, b.v,
                                                 (short)0, c, false, false);
}

// ---------------------------------------------------------------------------
// Tensor Data Mover: DMA a [rows x 32] bf16 tile (row stride = Kdim elements)
// from global memory into LDS at byte offset lds_off.  D# per CDNA5 ISA ch.8.
// ---------------------------------------------------------------------------
__device__ __forceinline__ void tdm_load_tile(unsigned lds_off,
                                              const unsigned short* gptr,
                                              int Kdim, int rows) {
  unsigned long long ga = (unsigned long long)(size_t)gptr;
  u32x4 g0;
  g0[0] = 1u;                                   // count=1, user descriptor
  g0[1] = lds_off;                              // lds_addr (bytes)
  g0[2] = (unsigned)ga;                         // global_addr[31:0]
  g0[3] = (unsigned)((ga >> 32) & 0x01FFFFFFull) | (2u << 30);  // ga[56:32] | type=2
  unsigned K = (unsigned)Kdim, R = (unsigned)rows;
  i32x8 g1;
  g1[0] = 0x00010000;                           // wg_mask=0, data_size=2 bytes
  g1[1] = (int)((K & 0xFFFFu) << 16);           // atomic_addr=0 | tensor_dim0 lo16
  g1[2] = (int)(((K >> 16) & 0xFFFFu) | ((R & 0xFFFFu) << 16));   // d0 hi | d1 lo
  g1[3] = (int)(((R >> 16) & 0xFFFFu) | (32u << 16));             // d1 hi | tile_dim0=32
  g1[4] = (int)(R & 0xFFFFu);                   // tile_dim1=rows, tile_dim2=0
  g1[5] = (int)K;                               // tensor_dim0_stride lo32
  g1[6] = 0;                                    // stride hi | dim1_stride lo
  g1[7] = 0;
  i32x4 z4 = {0, 0, 0, 0};
#if defined(__clang_major__) && __clang_major__ >= 23
  i32x8 z8 = {0, 0, 0, 0, 0, 0, 0, 0};
  __builtin_amdgcn_tensor_load_to_lds(g0, g1, z4, z4, z8, 0);
#else
  __builtin_amdgcn_tensor_load_to_lds(g0, g1, z4, z4, 0);
#endif
}

// ---------------------------------------------------------------------------
// fp32 -> bf16 bulk convert
// ---------------------------------------------------------------------------
__global__ __launch_bounds__(256) void cvt_bf16_kernel(const float* __restrict__ src,
                                                       unsigned short* __restrict__ dst,
                                                       int n) {
  int i = blockIdx.x * 256 + threadIdx.x;
  if (i < n) dst[i] = f2bf(src[i]);
}

// ---------------------------------------------------------------------------
// viewmats: M = inv(c2w) (rigid), Minv = c2w ; views = [ref0, ref1, tar]
// ---------------------------------------------------------------------------
__global__ void prep_mats_kernel(const float* __restrict__ ref_c2w,
                                 const float* __restrict__ tar_c2w,
                                 float* __restrict__ Mv, float* __restrict__ Mi) {
  if (threadIdx.x != 0 || blockIdx.x != 0) return;
  for (int v = 0; v < 3; ++v) {
    const float* c = (v < 2) ? (ref_c2w + v * 16) : tar_c2w;
    for (int i = 0; i < 16; ++i) Mi[v * 16 + i] = c[i];
    float t0 = c[3], t1 = c[7], t2 = c[11];
    float* m = Mv + v * 16;
    m[0] = c[0]; m[1] = c[4]; m[2] = c[8];
    m[4] = c[1]; m[5] = c[5]; m[6] = c[9];
    m[8] = c[2]; m[9] = c[6]; m[10] = c[10];
    m[3]  = -(c[0] * t0 + c[4] * t1 + c[8]  * t2);
    m[7]  = -(c[1] * t0 + c[5] * t1 + c[9]  * t2);
    m[11] = -(c[2] * t0 + c[6] * t1 + c[10] * t2);
    m[12] = 0.f; m[13] = 0.f; m[14] = 0.f; m[15] = 1.f;
  }
}

// ---------------------------------------------------------------------------
// Plucker rays + patchify -> bf16 feature matrices
// feat_ref [2048, 576] : [img(192) | rays(384)] ; feat_tar [1024, 384]
// ---------------------------------------------------------------------------
__global__ __launch_bounds__(256) void rays_feat_kernel(
    const float* __restrict__ imgs, const float* __restrict__ ref_c2w,
    const float* __restrict__ ref_K, const float* __restrict__ tar_c2w,
    const float* __restrict__ tar_K, unsigned short* __restrict__ fr,
    unsigned short* __restrict__ ft) {
  int gid = blockIdx.x * 256 + threadIdx.x;
  if (gid >= 3 * IMG * IMG) return;
  int v = gid / (IMG * IMG);
  int p = gid - v * (IMG * IMG);
  int h = p / IMG, w = p - h * IMG;
  const float* c2w = (v < 2) ? (ref_c2w + v * 16) : tar_c2w;
  const float* Km = (v < 2) ? (ref_K + v * 9) : tar_K;
  float a = Km[0], b = Km[1], c = Km[2], d = Km[3], e = Km[4], f = Km[5];
  float g = Km[6], hh = Km[7], i9 = Km[8];
  float A = e * i9 - f * hh, Bc = -(d * i9 - f * g), C = d * hh - e * g;
  float id = 1.f / (a * A + b * Bc + c * C);
  float k00 = A * id, k01 = -(b * i9 - c * hh) * id, k02 = (b * f - c * e) * id;
  float k10 = Bc * id, k11 = (a * i9 - c * g) * id, k12 = -(a * f - c * d) * id;
  float k20 = C * id, k21 = -(a * hh - b * g) * id, k22 = (a * e - b * d) * id;
  float u = (float)w + 0.5f, vv = (float)h + 0.5f;
  float dc0 = k00 * u + k01 * vv + k02;
  float dc1 = k10 * u + k11 * vv + k12;
  float dc2 = k20 * u + k21 * vv + k22;
  float dx = c2w[0] * dc0 + c2w[1] * dc1 + c2w[2] * dc2;
  float dy = c2w[4] * dc0 + c2w[5] * dc1 + c2w[6] * dc2;
  float dz = c2w[8] * dc0 + c2w[9] * dc1 + c2w[10] * dc2;
  float invn = rsqrtf(dx * dx + dy * dy + dz * dz);
  dx *= invn; dy *= invn; dz *= invn;
  float ox = c2w[3], oy = c2w[7], oz = c2w[11];
  float mx = oy * dz - oz * dy, my = oz * dx - ox * dz, mz = ox * dy - oy * dx;
  int tok = (h >> 3) * 32 + (w >> 3);
  int py = h & 7, px = w & 7;
  if (v < 2) {
    size_t base = (size_t)(v * NTOK + tok) * 576;
    int ci = (py * 8 + px) * 3;
    const float* ip = imgs + (((size_t)v * IMG + h) * IMG + w) * 3;
    fr[base + ci + 0] = f2bf(ip[0]);
    fr[base + ci + 1] = f2bf(ip[1]);
    fr[base + ci + 2] = f2bf(ip[2]);
    int ri = 192 + (py * 8 + px) * 6;
    fr[base + ri + 0] = f2bf(dx); fr[base + ri + 1] = f2bf(dy);
    fr[base + ri + 2] = f2bf(dz); fr[base + ri + 3] = f2bf(mx);
    fr[base + ri + 4] = f2bf(my); fr[base + ri + 5] = f2bf(mz);
  } else {
    size_t base = (size_t)tok * 384;
    int ri = (py * 8 + px) * 6;
    ft[base + ri + 0] = f2bf(dx); ft[base + ri + 1] = f2bf(dy);
    ft[base + ri + 2] = f2bf(dz); ft[base + ri + 3] = f2bf(mx);
    ft[base + ri + 4] = f2bf(my); ft[base + ri + 5] = f2bf(mz);
  }
}

// ---------------------------------------------------------------------------
// LayerNorm (row-wise over D=768). OUT_BF=1 -> bf16 out, else fp32 (in-place OK)
// ---------------------------------------------------------------------------
template <int OUT_BF>
__global__ __launch_bounds__(256) void layernorm_kernel(
    const float* __restrict__ x, const float* __restrict__ g,
    float* __restrict__ of, unsigned short* __restrict__ ob) {
  __shared__ float buf[256];
  int row = blockIdx.x, tid = threadIdx.x;
  const float* xr = x + (size_t)row * DM;
  float s = 0.f;
  for (int i = tid; i < DM; i += 256) s += xr[i];
  buf[tid] = s; __syncthreads();
  for (int st = 128; st > 0; st >>= 1) {
    if (tid < st) buf[tid] += buf[tid + st];
    __syncthreads();
  }
  float mu = buf[0] * (1.f / DM);
  __syncthreads();
  float vs = 0.f;
  for (int i = tid; i < DM; i += 256) { float dd = xr[i] - mu; vs += dd * dd; }
  buf[tid] = vs; __syncthreads();
  for (int st = 128; st > 0; st >>= 1) {
    if (tid < st) buf[tid] += buf[tid + st];
    __syncthreads();
  }
  float inv = rsqrtf(buf[0] * (1.f / DM) + 1e-5f);
  for (int i = tid; i < DM; i += 256) {
    float y = (xr[i] - mu) * inv * g[i];
    if (OUT_BF) ob[(size_t)row * DM + i] = f2bf(y);
    else        of[(size_t)row * DM + i] = y;
  }
}

// ---------------------------------------------------------------------------
// WMMA GEMM: C[M,N] = A[M,K] @ B[N,K]^T, A/B bf16, contiguous K.
// Block 128 thr (4 waves), block tile 128x64, wave = 32 rows x 64 cols.
// LDS tiles DMA'd by the Tensor Data Mover, double buffered (TENSORcnt).
// MODE 0: Cf = G ; MODE 1: Cf += G ; MODE 2: Cbf = bf16(relu(G))
// M multiple of 128; N multiple of 64; K multiple of 32.
// ---------------------------------------------------------------------------
template <int MODE>
__global__ __launch_bounds__(128) void gemm_bf16_kernel(
    const unsigned short* __restrict__ A, const unsigned short* __restrict__ B,
    float* __restrict__ Cf, unsigned short* __restrict__ Cbf, int M, int N, int K) {
  __shared__ alignas(16) unsigned short sA[2][128 * 32];
  __shared__ alignas(16) unsigned short sB[2][64 * 32];
  const int tid = threadIdx.x;
  const int lane = tid & 31;
  const int wave = tid >> 5;
  const int bm = blockIdx.x * 128;
  const int bn = blockIdx.y * 64;
  const int nk = K >> 5;
  v8f acc[2][4];
#pragma unroll
  for (int m = 0; m < 2; ++m)
#pragma unroll
    for (int n = 0; n < 4; ++n) acc[m][n] = vzero8();

  const unsigned ldsA0 = (unsigned)(size_t)(void*)&sA[0][0];
  const unsigned ldsA1 = (unsigned)(size_t)(void*)&sA[1][0];
  const unsigned ldsB0 = (unsigned)(size_t)(void*)&sB[0][0];
  const unsigned ldsB1 = (unsigned)(size_t)(void*)&sB[1][0];

  if (wave == 0) {
    tdm_load_tile(ldsA0, A + (size_t)bm * K, K, 128);
    tdm_load_tile(ldsB0, B + (size_t)bn * K, K, 64);
  }
  for (int i = 0; i < nk; ++i) {
    const int cur = i & 1;
    if (wave == 0) {
      if (i + 1 < nk) {
        const int k1 = (i + 1) * 32;
        tdm_load_tile(cur ? ldsA0 : ldsA1, A + (size_t)bm * K + k1, K, 128);
        tdm_load_tile(cur ? ldsB0 : ldsB1, B + (size_t)bn * K + k1, K, 64);
        __builtin_amdgcn_s_wait_tensorcnt(2);   // tile i complete (in-order)
      } else {
        __builtin_amdgcn_s_wait_tensorcnt(0);
      }
    }
    __syncthreads();                            // publish tile i to all waves
    const unsigned short* pA = sA[cur];
    const unsigned short* pB = sB[cur];
    FragBF fa[2];
    {
      const int hb = (lane & 16) ? 8 : 0;
#pragma unroll
      for (int mt = 0; mt < 2; ++mt) {
        int row = wave * 32 + mt * 16 + (lane & 15);
        fa[mt].q[0] = *reinterpret_cast<const uint4*>(&pA[row * 32 + hb]);
        fa[mt].q[1] = *reinterpret_cast<const uint4*>(&pA[row * 32 + hb + 16]);
      }
    }
#pragma unroll
    for (int nt = 0; nt < 4; ++nt) {
      FragBF fb;
      int col = nt * 16 + (lane & 15);
      int kh = (lane & 16) ? 8 : 0;
      fb.q[0] = *reinterpret_cast<const uint4*>(&pB[col * 32 + kh]);
      fb.q[1] = *reinterpret_cast<const uint4*>(&pB[col * 32 + kh + 16]);
#pragma unroll
      for (int mt = 0; mt < 2; ++mt) acc[mt][nt] = wmma_bf(fa[mt], fb, acc[mt][nt]);
    }
    __syncthreads();                            // all reads done before overwrite
  }
  const int n0 = lane & 15;
  const int rb = (lane & 16) ? 8 : 0;
#pragma unroll
  for (int mt = 0; mt < 2; ++mt)
#pragma unroll
    for (int nt = 0; nt < 4; ++nt)
#pragma unroll
      for (int r = 0; r < 8; ++r) {
        int gr = bm + wave * 32 + mt * 16 + rb + r;
        int gc = bn + nt * 16 + n0;
        size_t idx = (size_t)gr * N + gc;
        float vvv = acc[mt][nt][r];
        if (MODE == 0)      Cf[idx] = vvv;
        else if (MODE == 1) Cf[idx] += vvv;
        else                Cbf[idx] = f2bf(vvv > 0.f ? vvv : 0.f);
      }
}

// ---------------------------------------------------------------------------
// Fused flash attention with per-token SE(3) transforms.
// One wave per (16-query tile, head). qkv fp32 [SEQ,2304]; obf bf16 [SEQ,768].
// ---------------------------------------------------------------------------
__global__ __launch_bounds__(32) void attn_kernel(
    const float* __restrict__ qkv, const float* __restrict__ Mv,
    const float* __restrict__ Mi, unsigned short* __restrict__ obf) {
  const int qblk = blockIdx.x;   // 0..191
  const int head = blockIdx.y;   // 0..15
  const int lane = threadIdx.x;
  __shared__ alignas(16) unsigned short lq[16 * 64];
  __shared__ alignas(16) unsigned short lk[32 * 64];
  __shared__ alignas(16) unsigned short lvT[64 * 32];
  __shared__ alignas(16) unsigned short lp[16 * 32];
  __shared__ float lo[16 * 64];
  __shared__ float linv[16];

  for (int i = lane; i < 16 * 64; i += 32) lq[i] = 0;
  for (int i = lane; i < 32 * 64; i += 32) lk[i] = 0;
  for (int i = lane; i < 64 * 32; i += 32) lvT[i] = 0;

  const float scale = 0.14433756729740643f;  // 1/sqrt(48)
  for (int task = lane; task < 192; task += 32) {
    int tq = task / 12, g = task - tq * 12;
    int tok = qblk * 16 + tq;
    const float* M = Mv + (tok >> 10) * 16;
    const float* src = qkv + (size_t)tok * 2304 + head * HDIM + g * 4;
    float x0 = src[0], x1 = src[1], x2 = src[2], x3 = src[3];
    float y0 = M[0] * x0 + M[1] * x1 + M[2] * x2 + M[3] * x3;
    float y1 = M[4] * x0 + M[5] * x1 + M[6] * x2 + M[7] * x3;
    float y2 = M[8] * x0 + M[9] * x1 + M[10] * x2 + M[11] * x3;
    float y3 = M[12] * x0 + M[13] * x1 + M[14] * x2 + M[15] * x3;
    int off = tq * 64 + g * 4;
    lq[off + 0] = f2bf(y0 * scale); lq[off + 1] = f2bf(y1 * scale);
    lq[off + 2] = f2bf(y2 * scale); lq[off + 3] = f2bf(y3 * scale);
  }
  __syncthreads();
  FragBF qa0, qa1;
  {
    int row = lane & 15;
    int hb = (lane & 16) ? 8 : 0;
    qa0.q[0] = *reinterpret_cast<const uint4*>(&lq[row * 64 + hb]);
    qa0.q[1] = *reinterpret_cast<const uint4*>(&lq[row * 64 + hb + 16]);
    qa1.q[0] = *reinterpret_cast<const uint4*>(&lq[row * 64 + 32 + hb]);
    qa1.q[1] = *reinterpret_cast<const uint4*>(&lq[row * 64 + 32 + hb + 16]);
  }
  v8f oacc[4];
#pragma unroll
  for (int i = 0; i < 4; ++i) oacc[i] = vzero8();
  float mrow[8], lrow[8];
#pragma unroll
  for (int r = 0; r < 8; ++r) { mrow[r] = -1e30f; lrow[r] = 0.f; }

  for (int kb = 0; kb < SEQ; kb += 32) {
    __syncthreads();
    for (int task = lane; task < 384; task += 32) {
      int tk = task / 12, g = task - tk * 12;
      int tok = kb + tk;
      const float* M = Mv + (tok >> 10) * 16;
      const float* ks = qkv + (size_t)tok * 2304 + DM + head * HDIM + g * 4;
      const float* vs = ks + DM;
      float x0 = ks[0], x1 = ks[1], x2 = ks[2], x3 = ks[3];
      float y0 = M[0] * x0 + M[1] * x1 + M[2] * x2 + M[3] * x3;
      float y1 = M[4] * x0 + M[5] * x1 + M[6] * x2 + M[7] * x3;
      float y2 = M[8] * x0 + M[9] * x1 + M[10] * x2 + M[11] * x3;
      float y3 = M[12] * x0 + M[13] * x1 + M[14] * x2 + M[15] * x3;
      int ko = tk * 64 + g * 4;
      lk[ko + 0] = f2bf(y0); lk[ko + 1] = f2bf(y1);
      lk[ko + 2] = f2bf(y2); lk[ko + 3] = f2bf(y3);
      float z0 = vs[0], z1 = vs[1], z2 = vs[2], z3 = vs[3];
      float w0 = M[0] * z0 + M[1] * z1 + M[2] * z2 + M[3] * z3;
      float w1 = M[4] * z0 + M[5] * z1 + M[6] * z2 + M[7] * z3;
      float w2 = M[8] * z0 + M[9] * z1 + M[10] * z2 + M[11] * z3;
      float w3 = M[12] * z0 + M[13] * z1 + M[14] * z2 + M[15] * z3;
      lvT[(g * 4 + 0) * 32 + tk] = f2bf(w0);
      lvT[(g * 4 + 1) * 32 + tk] = f2bf(w1);
      lvT[(g * 4 + 2) * 32 + tk] = f2bf(w2);
      lvT[(g * 4 + 3) * 32 + tk] = f2bf(w3);
    }
    __syncthreads();
    v8f s0 = vzero8(), s1 = vzero8();
    {
      int kh = (lane & 16) ? 8 : 0;
      int key0 = lane & 15;
      FragBF b;
      b.q[0] = *reinterpret_cast<const uint4*>(&lk[key0 * 64 + kh]);
      b.q[1] = *reinterpret_cast<const uint4*>(&lk[key0 * 64 + kh + 16]);
      s0 = wmma_bf(qa0, b, s0);
      b.q[0] = *reinterpret_cast<const uint4*>(&lk[key0 * 64 + 32 + kh]);
      b.q[1] = *reinterpret_cast<const uint4*>(&lk[key0 * 64 + 32 + kh + 16]);
      s0 = wmma_bf(qa1, b, s0);
      int key1 = 16 + key0;
      b.q[0] = *reinterpret_cast<const uint4*>(&lk[key1 * 64 + kh]);
      b.q[1] = *reinterpret_cast<const uint4*>(&lk[key1 * 64 + kh + 16]);
      s1 = wmma_bf(qa0, b, s1);
      b.q[0] = *reinterpret_cast<const uint4*>(&lk[key1 * 64 + 32 + kh]);
      b.q[1] = *reinterpret_cast<const uint4*>(&lk[key1 * 64 + 32 + kh + 16]);
      s1 = wmma_bf(qa1, b, s1);
    }
#pragma unroll
    for (int r = 0; r < 8; ++r) {
      float t0 = s0[r], t1 = s1[r];
      float tm = fmaxf(t0, t1);
      for (int off = 8; off >= 1; off >>= 1) tm = fmaxf(tm, __shfl_xor(tm, off, 32));
      float mn = fmaxf(mrow[r], tm);
      float alpha = __expf(mrow[r] - mn);
      float p0 = __expf(t0 - mn), p1 = __expf(t1 - mn);
      float ps = p0 + p1;
      for (int off = 8; off >= 1; off >>= 1) ps += __shfl_xor(ps, off, 32);
      lrow[r] = lrow[r] * alpha + ps;
      mrow[r] = mn;
#pragma unroll
      for (int nt = 0; nt < 4; ++nt) oacc[nt][r] *= alpha;
      int row = ((lane & 16) ? 8 : 0) + r;
      lp[row * 32 + (lane & 15)] = f2bf(p0);
      lp[row * 32 + 16 + (lane & 15)] = f2bf(p1);
    }
    __syncthreads();
    FragBF pf;
    {
      int row = lane & 15;
      int hb = (lane & 16) ? 8 : 0;
      pf.q[0] = *reinterpret_cast<const uint4*>(&lp[row * 32 + hb]);
      pf.q[1] = *reinterpret_cast<const uint4*>(&lp[row * 32 + hb + 16]);
    }
    {
      int dimc = lane & 15;
      int kh = (lane & 16) ? 8 : 0;
#pragma unroll
      for (int nt = 0; nt < 4; ++nt) {
        FragBF bv;
        int dim = nt * 16 + dimc;
        bv.q[0] = *reinterpret_cast<const uint4*>(&lvT[dim * 32 + kh]);
        bv.q[1] = *reinterpret_cast<const uint4*>(&lvT[dim * 32 + kh + 16]);
        oacc[nt] = wmma_bf(pf, bv, oacc[nt]);
      }
    }
  }
  __syncthreads();
  {
    int rb = (lane & 16) ? 8 : 0;
#pragma unroll
    for (int nt = 0; nt < 4; ++nt)
#pragma unroll
      for (int r = 0; r < 8; ++r)
        lo[(rb + r) * 64 + nt * 16 + (lane & 15)] = oacc[nt][r];
    if ((lane & 15) == 0)
      for (int r = 0; r < 8; ++r) linv[((lane & 16) ? 8 : 0) + r] = 1.f / lrow[r];
  }
  __syncthreads();
  for (int task = lane; task < 192; task += 32) {
    int tq = task / 12, g = task - tq * 12;
    int tok = qblk * 16 + tq;
    const float* M = Mi + (tok >> 10) * 16;
    float il = linv[tq];
    float x0 = lo[tq * 64 + g * 4 + 0] * il;
    float x1 = lo[tq * 64 + g * 4 + 1] * il;
    float x2 = lo[tq * 64 + g * 4 + 2] * il;
    float x3 = lo[tq * 64 + g * 4 + 3] * il;
    float y0 = M[0] * x0 + M[1] * x1 + M[2] * x2 + M[3] * x3;
    float y1 = M[4] * x0 + M[5] * x1 + M[6] * x2 + M[7] * x3;
    float y2 = M[8] * x0 + M[9] * x1 + M[10] * x2 + M[11] * x3;
    float y3 = M[12] * x0 + M[13] * x1 + M[14] * x2 + M[15] * x3;
    unsigned short* dst = obf + (size_t)tok * DM + head * HDIM + g * 4;
    dst[0] = f2bf(y0); dst[1] = f2bf(y1); dst[2] = f2bf(y2); dst[3] = f2bf(y3);
  }
}

// ---------------------------------------------------------------------------
// unpatchify [1024,192] -> [256,256,3]
// ---------------------------------------------------------------------------
__global__ __launch_bounds__(256) void unpatchify_kernel(const float* __restrict__ t,
                                                         float* __restrict__ out) {
  int gid = blockIdx.x * 256 + threadIdx.x;
  if (gid >= IMG * IMG * 3) return;
  int c = gid % 3;
  int w = (gid / 3) % IMG;
  int h = gid / (3 * IMG);
  int n = (h >> 3) * 32 + (w >> 3);
  int ch = ((h & 7) * 8 + (w & 7)) * 3 + c;
  out[gid] = t[(size_t)n * 192 + ch];
}

// ---------------------------------------------------------------------------
extern "C" void kernel_launch(void* const* d_in, const int* in_sizes, int n_in,
                              void* d_out, int out_size, void* d_ws, size_t ws_size,
                              hipStream_t stream) {
  (void)in_sizes; (void)n_in; (void)out_size; (void)ws_size;
  const float* ref_imgs = (const float*)d_in[0];
  const float* ref_c2w  = (const float*)d_in[1];
  const float* ref_K    = (const float*)d_in[2];
  const float* tar_c2w  = (const float*)d_in[3];
  const float* tar_K    = (const float*)d_in[4];
  const float* W_in  = (const float*)d_in[5];
  const float* W_q   = (const float*)d_in[6];
  const float* W_out = (const float*)d_in[7];
  const float* gin   = (const float*)d_in[8];
  const float* gout  = (const float*)d_in[9];
  const float* Wqkv  = (const float*)d_in[10];
  const float* Wo    = (const float*)d_in[11];
  const float* g1    = (const float*)d_in[12];
  const float* g2    = (const float*)d_in[13];
  const float* W1    = (const float*)d_in[14];
  const float* W2    = (const float*)d_in[15];
  float* out = (float*)d_out;
  char* ws = (char*)d_ws;

  size_t off = 0;
  auto take = [&](size_t bytes) -> char* {
    char* p = ws + off;
    off = (off + bytes + 255) & ~(size_t)255;
    return p;
  };
  unsigned short* Winb  = (unsigned short*)take((size_t)DM * 576 * 2);
  unsigned short* Wqb   = (unsigned short*)take((size_t)DM * 384 * 2);
  unsigned short* Wqkvb = (unsigned short*)take((size_t)NLAY * 3 * DM * DM * 2);
  unsigned short* Wob   = (unsigned short*)take((size_t)NLAY * DM * DM * 2);
  unsigned short* W1b   = (unsigned short*)take((size_t)NLAY * DFFN * DM * 2);
  unsigned short* W2b   = (unsigned short*)take((size_t)NLAY * DM * DFFN * 2);
  unsigned short* Woutb = (unsigned short*)take((size_t)192 * DM * 2);
  float* Mv = (float*)take(48 * 4);
  float* Mi = (float*)take(48 * 4);
  unsigned short* featR = (unsigned short*)take((size_t)2048 * 576 * 2);
  unsigned short* featT = (unsigned short*)take((size_t)1024 * 384 * 2);
  float* x = (float*)take((size_t)SEQ * DM * 4);
  unsigned short* hbf = (unsigned short*)take((size_t)SEQ * DM * 2);
  float* qkvb = (float*)take((size_t)SEQ * 3 * DM * 4);
  unsigned short* obf = (unsigned short*)take((size_t)SEQ * DM * 2);
  unsigned short* ffb = (unsigned short*)take((size_t)SEQ * DFFN * 2);
  float* tmpo = (float*)take((size_t)1024 * 192 * 4);

  auto cvt = [&](const float* src, unsigned short* dst, size_t n) {
    cvt_bf16_kernel<<<dim3((unsigned)((n + 255) / 256)), dim3(256), 0, stream>>>(src, dst, (int)n);
  };
  cvt(W_in,  Winb,  (size_t)DM * 576);
  cvt(W_q,   Wqb,   (size_t)DM * 384);
  cvt(Wqkv,  Wqkvb, (size_t)NLAY * 3 * DM * DM);
  cvt(Wo,    Wob,   (size_t)NLAY * DM * DM);
  cvt(W1,    W1b,   (size_t)NLAY * DFFN * DM);
  cvt(W2,    W2b,   (size_t)NLAY * DM * DFFN);
  cvt(W_out, Woutb, (size_t)192 * DM);

  prep_mats_kernel<<<1, 1, 0, stream>>>(ref_c2w, tar_c2w, Mv, Mi);
  rays_feat_kernel<<<dim3(3 * IMG * IMG / 256), dim3(256), 0, stream>>>(
      ref_imgs, ref_c2w, ref_K, tar_c2w, tar_K, featR, featT);

  gemm_bf16_kernel<0><<<dim3(2048 / 128, DM / 64), dim3(128), 0, stream>>>(
      featR, Winb, x, nullptr, 2048, DM, 576);
  gemm_bf16_kernel<0><<<dim3(1024 / 128, DM / 64), dim3(128), 0, stream>>>(
      featT, Wqb, x + (size_t)2048 * DM, nullptr, 1024, DM, 384);
  layernorm_kernel<0><<<dim3(SEQ), dim3(256), 0, stream>>>(x, gin, x, nullptr);

  for (int l = 0; l < NLAY; ++l) {
    const unsigned short* wqkv_l = Wqkvb + (size_t)l * 3 * DM * DM;
    const unsigned short* wo_l   = Wob   + (size_t)l * DM * DM;
    const unsigned short* w1_l   = W1b   + (size_t)l * DFFN * DM;
    const unsigned short* w2_l   = W2b   + (size_t)l * DM * DFFN;
    layernorm_kernel<1><<<dim3(SEQ), dim3(256), 0, stream>>>(x, g1 + (size_t)l * DM, nullptr, hbf);
    gemm_bf16_kernel<0><<<dim3(SEQ / 128, 3 * DM / 64), dim3(128), 0, stream>>>(
        hbf, wqkv_l, qkvb, nullptr, SEQ, 3 * DM, DM);
    attn_kernel<<<dim3(SEQ / 16, NHEAD), dim3(32), 0, stream>>>(qkvb, Mv, Mi, obf);
    gemm_bf16_kernel<1><<<dim3(SEQ / 128, DM / 64), dim3(128), 0, stream>>>(
        obf, wo_l, x, nullptr, SEQ, DM, DM);
    layernorm_kernel<1><<<dim3(SEQ), dim3(256), 0, stream>>>(x, g2 + (size_t)l * DM, nullptr, hbf);
    gemm_bf16_kernel<2><<<dim3(SEQ / 128, DFFN / 64), dim3(128), 0, stream>>>(
        hbf, w1_l, nullptr, ffb, SEQ, DFFN, DM);
    gemm_bf16_kernel<1><<<dim3(SEQ / 128, DM / 64), dim3(128), 0, stream>>>(
        ffb, w2_l, x, nullptr, SEQ, DM, DFFN);
  }

  layernorm_kernel<1><<<dim3(1024), dim3(256), 0, stream>>>(
      x + (size_t)2048 * DM, gout, nullptr, hbf);
  gemm_bf16_kernel<0><<<dim3(1024 / 128, 192 / 64), dim3(128), 0, stream>>>(
      hbf, Woutb, tmpo, nullptr, 1024, 192, DM);
  unpatchify_kernel<<<dim3((IMG * IMG * 3 + 255) / 256), dim3(256), 0, stream>>>(tmpo, out);
}
